// ScratchpadPrunedAttention_19095424598884
// MI455X (gfx1250) — compile-verified, matrix operand order checked by reading, not run on hardware
//
#include <hip/hip_runtime.h>
#include <hip/hip_bf16.h>
#include <cstdint>

// ---------------- types ----------------
typedef __bf16 bf16_t;
typedef __attribute__((ext_vector_type(16))) __bf16 v16bf;
typedef __attribute__((ext_vector_type(8)))  __bf16 v8bf;
typedef __attribute__((ext_vector_type(8)))  float  v8f;

#define WMMA_BF16(a, b, c) \
  __builtin_amdgcn_wmma_f32_16x16x32_bf16(false, (a), false, (b), (short)0, (c), false, false)

// problem constants
constexpr int Bc   = 2;
constexpr int Tc   = 2048;
constexpr int Cc   = 2048;
constexpr int Hc   = 16;
constexpr int HKVc = 4;
constexpr int Dc   = 128;
constexpr int GRP  = Hc / HKVc;

// fp32 -> bf16, round-to-nearest-even
__device__ __forceinline__ bf16_t f2bf(float f) {
  union { float f; uint32_t u; } x; x.f = f;
  uint32_t r = x.u + 0x7fffu + ((x.u >> 16) & 1u);
  uint16_t h = (uint16_t)(r >> 16);
  return __builtin_bit_cast(bf16_t, h);
}

// WMMA fragment load from an LDS row of 32 contiguous K elements (ISA 7.12.2:
// lanes 0-15 hold K=0..7,16..23; lanes 16-31 hold K=8..15,24..31).
__device__ __forceinline__ v16bf frag_ld(const bf16_t* row32, int hi) {
  union { v16bf v; v8bf h[2]; } u;
  u.h[0] = *reinterpret_cast<const v8bf*>(row32 + hi * 8);
  u.h[1] = *reinterpret_cast<const v8bf*>(row32 + 16 + hi * 8);
  return u.v;
}

// ---- CDNA5 async global->LDS copy (ASYNCcnt path, CDNA5 ISA) ----
__device__ __forceinline__ void async_ld_b128(void* lds, const void* gsrc) {
  uint32_t la = (uint32_t)(uintptr_t)lds;  // low 32 bits of generic ptr = LDS offset
  uint64_t ga = (uint64_t)(uintptr_t)gsrc;
  asm volatile("global_load_async_to_lds_b128 %0, %1, off" ::"v"(la), "v"(ga)
               : "memory");
}
__device__ __forceinline__ void wait_async0() {
  asm volatile("s_wait_asynccnt 0x0" ::: "memory");
}

// ---------------- fp32 -> bf16 conversion ----------------
__global__ void cvt_f32_to_bf16_kernel(const float* __restrict__ src,
                                       bf16_t* __restrict__ dst, int n) {
  int i = blockIdx.x * blockDim.x + threadIdx.x;
  if (i < n) dst[i] = f2bf(src[i]);
}

// ---------------- generic WMMA bf16 GEMM: C[M,N] = A[M,K] @ B[K,N] ----------------
template <bool OUT_F32>
__global__ __launch_bounds__(256) void gemm_bf16_kernel(
    const bf16_t* __restrict__ A, const bf16_t* __restrict__ Bm,
    void* __restrict__ Cout, int M, int N, int K) {
  __shared__ __align__(16) bf16_t As[64][32];
  __shared__ __align__(16) bf16_t Bt[64][32];

  const int tid  = threadIdx.x;
  const int wave = tid >> 5, lane = tid & 31;
  const int hi   = lane >> 4, l15 = lane & 15;
  const int m0   = blockIdx.y * 64;
  const int n0   = blockIdx.x * 64;
  const int rowBand  = wave & 3;
  const int colBand0 = (wave >> 2) * 2;

  v8f acc0 = {}, acc1 = {};
  const int ar = tid >> 2;
  const int ac = (tid & 3) * 8;

  for (int k0 = 0; k0 < K; k0 += 32) {
    *reinterpret_cast<v8bf*>(&As[ar][ac]) =
        *reinterpret_cast<const v8bf*>(&A[(size_t)(m0 + ar) * K + k0 + ac]);
#pragma unroll
    for (int j = 0; j < 8; ++j) {
      int idx = tid + 256 * j;
      int n = idx & 63, kk = idx >> 6;
      Bt[n][kk] = Bm[(size_t)(k0 + kk) * N + n0 + n];
    }
    __syncthreads();
    v16bf a  = frag_ld(&As[rowBand * 16 + l15][0], hi);
    v16bf b0 = frag_ld(&Bt[colBand0 * 16 + l15][0], hi);
    v16bf b1 = frag_ld(&Bt[(colBand0 + 1) * 16 + l15][0], hi);
    acc0 = WMMA_BF16(a, b0, acc0);
    acc1 = WMMA_BF16(a, b1, acc1);
    __syncthreads();
  }

#pragma unroll
  for (int j = 0; j < 8; ++j) {
    int m   = m0 + rowBand * 16 + j + hi * 8;
    int n_a = n0 + colBand0 * 16 + l15;
    int n_b = n0 + (colBand0 + 1) * 16 + l15;
    if constexpr (OUT_F32) {
      float* Cf = (float*)Cout;
      Cf[(size_t)m * N + n_a] = acc0[j];
      Cf[(size_t)m * N + n_b] = acc1[j];
    } else {
      bf16_t* Cb = (bf16_t*)Cout;
      Cb[(size_t)m * N + n_a] = f2bf(acc0[j]);
      Cb[(size_t)m * N + n_b] = f2bf(acc1[j]);
    }
  }
}

// ---------------- fused attention: QK^T + softmax + prune + renorm ----------------
// Block = 32 query rows x full S. Pass A: WMMA + branchless online (m, Z, Zk) —
// scores never touch memory. Pass B: recompute tiles, write final weights once.
// K tiles stream through LDS via async global->LDS copies (K lives in L2).
// Each wave owns one 16-col band and both 16-row bands; the B fragment is reused
// across the two row-band WMMAs.
__global__ __launch_bounds__(256) void attn_fused_kernel(
    const bf16_t* __restrict__ Q,   // [B*T, H*D]
    const bf16_t* __restrict__ Kb,  // [B*T, HKV*D]
    const int* __restrict__ keep,   // [T]
    float* __restrict__ attn) {     // [B*H, T, T]
  const int t0  = blockIdx.x * 32;
  const int bh  = blockIdx.y;
  const int b   = bh / Hc, h = bh % Hc;
  const int hkv = h / GRP;
  const int tid = threadIdx.x;
  const int wave = tid >> 5, lane = tid & 31;
  const int hi = lane >> 4, l15 = lane & 15;
  float* arow = attn + (size_t)bh * Tc * Tc;

  __shared__ __align__(16) bf16_t Qs[32][128];
  __shared__ __align__(16) bf16_t Ks[128][128];
  __shared__ float keepf[Tc];
  __shared__ float redM[8][32], redZ[8][32], redZk[8][32];
  __shared__ float rowM[32], rowInv[32];

  for (int s = tid; s < Tc; s += 256) keepf[s] = keep[s] ? 1.0f : 0.0f;

  {  // stage Q tile (async): 32 rows x 128 halves = 512 x 16B chunks
#pragma unroll
    for (int j2 = 0; j2 < 2; ++j2) {
      int idx = tid + 256 * j2;
      int r = idx >> 4, cc = (idx & 15) * 8;
      async_ld_b128(&Qs[r][cc],
                    Q + (size_t)(b * Tc + t0 + r) * (Hc * Dc) + h * Dc + cc);
    }
  }

  const int nSt = (t0 + 32 + 127) / 128;     // s-tiles covering the causal span
  const float scale = 0.08838834764831845f;  // 1/sqrt(128)

  // per-lane online stats: rows bb*16 + hi*8 + j
  float mr[2][8], zr[2][8], zkr[2][8];
#pragma unroll
  for (int bb = 0; bb < 2; ++bb)
#pragma unroll
    for (int j = 0; j < 8; ++j) {
      mr[bb][j] = -3.0e38f; zr[bb][j] = 0.f; zkr[bb][j] = 0.f;
    }

  // ---- pass A ----
  for (int st = 0; st < nSt; ++st) {
    const int s0 = st * 128;
#pragma unroll
    for (int j2 = 0; j2 < 8; ++j2) {  // K tile: 2048 x 16B chunks
      int idx = tid + 256 * j2;
      int r = idx >> 4, cc = (idx & 15) * 8;
      async_ld_b128(&Ks[r][cc],
                    Kb + (size_t)(b * Tc + s0 + r) * (HKVc * Dc) + hkv * Dc + cc);
    }
    wait_async0();
    __syncthreads();
    v8f acc0 = {}, acc1 = {};
#pragma unroll
    for (int kc = 0; kc < 4; ++kc) {
      v16bf bq = frag_ld(&Ks[wave * 16 + l15][kc * 32], hi);  // reused twice
      v16bf a0 = frag_ld(&Qs[l15][kc * 32], hi);
      v16bf a1 = frag_ld(&Qs[16 + l15][kc * 32], hi);
      acc0 = WMMA_BF16(a0, bq, acc0);
      acc1 = WMMA_BF16(a1, bq, acc1);
    }
    const int s = s0 + wave * 16 + l15;
    const float kf = keepf[s];
#pragma unroll
    for (int bb = 0; bb < 2; ++bb) {
#pragma unroll
      for (int j = 0; j < 8; ++j) {
        const int t = t0 + bb * 16 + hi * 8 + j;
        const bool valid = (s <= t);
        float raw = (bb == 0) ? acc0[j] : acc1[j];
        float v  = valid ? raw * scale : -3.0e38f;   // branchless (cndmask)
        float mn = fmaxf(mr[bb][j], v);
        float rs = __expf(mr[bb][j] - mn);
        float e  = valid ? __expf(v - mn) : 0.0f;    // avoid exp(0) pollution
        zr[bb][j]  = zr[bb][j] * rs + e;
        zkr[bb][j] = zkr[bb][j] * rs + e * kf;
        mr[bb][j]  = mn;
      }
    }
    __syncthreads();
  }

  // merge across the 16 lanes that share each row (different s columns)
#pragma unroll
  for (int o = 1; o < 16; o <<= 1) {
#pragma unroll
    for (int bb = 0; bb < 2; ++bb)
#pragma unroll
      for (int j = 0; j < 8; ++j) {
        float m2  = __shfl_xor(mr[bb][j], o, 32);
        float z2  = __shfl_xor(zr[bb][j], o, 32);
        float zk2 = __shfl_xor(zkr[bb][j], o, 32);
        float mn  = fmaxf(mr[bb][j], m2);
        float a1 = __expf(mr[bb][j] - mn), a2 = __expf(m2 - mn);
        zr[bb][j]  = zr[bb][j] * a1 + z2 * a2;
        zkr[bb][j] = zkr[bb][j] * a1 + zk2 * a2;
        mr[bb][j]  = mn;
      }
  }
#pragma unroll
  for (int bb = 0; bb < 2; ++bb)
#pragma unroll
    for (int j = 0; j < 8; ++j) {  // one writer per row per wave
      if (l15 == j) {
        int r = bb * 16 + hi * 8 + j;
        redM[wave][r]  = mr[bb][j];
        redZ[wave][r]  = zr[bb][j];
        redZk[wave][r] = zkr[bb][j];
      }
    }
  __syncthreads();
  if (tid < 32) {  // merge across 8 waves; final row stats
    float m = redM[0][tid], z = redZ[0][tid], zk = redZk[0][tid];
    for (int w2 = 1; w2 < 8; ++w2) {
      float m2 = redM[w2][tid], z2 = redZ[w2][tid], zk2 = redZk[w2][tid];
      float mn = fmaxf(m, m2);
      float a1 = __expf(m - mn), a2 = __expf(m2 - mn);
      z  = z * a1 + z2 * a2;
      zk = zk * a1 + zk2 * a2;
      m  = mn;
    }
    rowM[tid]   = m;
    rowInv[tid] = 1.0f / (zk + 1e-6f * z);  // w = keep*e/(Zk + EPS*Z)
  }
  __syncthreads();

  // ---- pass B: recompute and write final weights once ----
  for (int st = 0; st < nSt; ++st) {
    const int s0 = st * 128;
#pragma unroll
    for (int j2 = 0; j2 < 8; ++j2) {
      int idx = tid + 256 * j2;
      int r = idx >> 4, cc = (idx & 15) * 8;
      async_ld_b128(&Ks[r][cc],
                    Kb + (size_t)(b * Tc + s0 + r) * (HKVc * Dc) + hkv * Dc + cc);
    }
    wait_async0();
    __syncthreads();
    v8f acc0 = {}, acc1 = {};
#pragma unroll
    for (int kc = 0; kc < 4; ++kc) {
      v16bf bq = frag_ld(&Ks[wave * 16 + l15][kc * 32], hi);
      v16bf a0 = frag_ld(&Qs[l15][kc * 32], hi);
      v16bf a1 = frag_ld(&Qs[16 + l15][kc * 32], hi);
      acc0 = WMMA_BF16(a0, bq, acc0);
      acc1 = WMMA_BF16(a1, bq, acc1);
    }
    const int s = s0 + wave * 16 + l15;
    const float kf = keepf[s];
#pragma unroll
    for (int bb = 0; bb < 2; ++bb) {
#pragma unroll
      for (int j = 0; j < 8; ++j) {
        const int r = bb * 16 + hi * 8 + j;
        const int t = t0 + r;
        float raw = (bb == 0) ? acc0[j] : acc1[j];
        float e = __expf(raw * scale - rowM[r]);
        float outv = (s <= t) ? kf * e * rowInv[r] : 0.0f;  // branchless
        arow[(size_t)t * Tc + s] = outv;
      }
    }
    __syncthreads();
  }

  // zero-fill uncovered region s in [nSt*128, Tc)
  const int sz0 = nSt * 128;
  if (sz0 < Tc) {
    const int wcols = Tc - sz0;
    for (int i = tid; i < 32 * wcols; i += 256) {
      int r = i / wcols, c = i % wcols;
      arow[(size_t)(t0 + r) * Tc + sz0 + c] = 0.0f;
    }
  }
}

// ---------------- context = attn_w @ V (causally truncated k-loop) ----------------
__global__ __launch_bounds__(256) void context_kernel(
    const float* __restrict__ attn,  // [B*H, T, T]
    const bf16_t* __restrict__ V,    // [B*T, HKV*D]
    bf16_t* __restrict__ ctx) {      // [B*T, H*D]
  const int t0  = blockIdx.y * 64;
  const int bh  = blockIdx.z;
  const int b   = bh / Hc, h = bh % Hc;
  const int hkv = h / GRP;
  const int tid = threadIdx.x;
  const float* arow = attn + (size_t)bh * Tc * Tc;

  __shared__ __align__(16) bf16_t As[64][32];
  __shared__ __align__(16) bf16_t Vt[128][32];

  const int wave = tid >> 5, lane = tid & 31;
  const int hi = lane >> 4, l15 = lane & 15;
  const int rowBand = wave & 3;
  const int colBase = (wave >> 2) * 4;

  v8f acc[4] = {};
  const int smax = t0 + 64;  // attn == 0 beyond the diagonal
  const int ar = tid >> 2, ac = (tid & 3) * 8;

  for (int ks = 0; ks < smax; ks += 32) {
    const float* asrc = &arow[(size_t)(t0 + ar) * Tc + ks + ac];
#pragma unroll
    for (int j = 0; j < 8; ++j) As[ar][ac + j] = f2bf(asrc[j]);
    if (ks + 32 < smax) __builtin_prefetch(asrc + 32, 0, 0);  // global_prefetch_b8
#pragma unroll
    for (int j = 0; j < 16; ++j) {
      int idx = tid + 256 * j;
      int d = idx & 127, ss = idx >> 7;
      Vt[d][ss] = V[(size_t)(b * Tc + ks + ss) * (HKVc * Dc) + hkv * Dc + d];
    }
    __syncthreads();
    v16bf a = frag_ld(&As[rowBand * 16 + l15][0], hi);
#pragma unroll
    for (int cb = 0; cb < 4; ++cb) {
      v16bf bv = frag_ld(&Vt[(colBase + cb) * 16 + l15][0], hi);
      acc[cb] = WMMA_BF16(a, bv, acc[cb]);
    }
    __syncthreads();
  }

#pragma unroll
  for (int cb = 0; cb < 4; ++cb) {
    int d = (colBase + cb) * 16 + l15;
#pragma unroll
    for (int j = 0; j < 8; ++j) {
      int t = t0 + rowBand * 16 + j + hi * 8;
      ctx[(size_t)(b * Tc + t) * (Hc * Dc) + h * Dc + d] = f2bf(acc[cb][j]);
    }
  }
}

// ---------------- launch ----------------
extern "C" void kernel_launch(void* const* d_in, const int* in_sizes, int n_in,
                              void* d_out, int out_size, void* d_ws, size_t ws_size,
                              hipStream_t stream) {
  const float* hs = (const float*)d_in[0];
  const float* Wq = (const float*)d_in[1];
  const float* Wk = (const float*)d_in[2];
  const float* Wv = (const float*)d_in[3];
  const float* Wo = (const float*)d_in[4];
  const int* keep = (const int*)d_in[5];

  float* out  = (float*)d_out;
  float* attn = out + (size_t)Bc * Tc * Cc;

  char* w = (char*)d_ws;
  auto carve = [&](size_t bytes) -> char* {
    char* p = w;
    w += (bytes + 255) & ~(size_t)255;
    return p;
  };
  const size_t nX = (size_t)Bc * Tc * Cc;
  const size_t nQ = (size_t)Bc * Tc * Hc * Dc;
  const size_t nK = (size_t)Bc * Tc * HKVc * Dc;
  bf16_t* Xb  = (bf16_t*)carve(nX * 2);
  bf16_t* Wqb = (bf16_t*)carve((size_t)Cc * Hc * Dc * 2);
  bf16_t* Wkb = (bf16_t*)carve((size_t)Cc * HKVc * Dc * 2);
  bf16_t* Wvb = (bf16_t*)carve((size_t)Cc * HKVc * Dc * 2);
  bf16_t* Wob = (bf16_t*)carve((size_t)Hc * Dc * Cc * 2);
  bf16_t* Qb  = (bf16_t*)carve(nQ * 2);
  bf16_t* Kbf = (bf16_t*)carve(nK * 2);
  bf16_t* Vbf = (bf16_t*)carve(nK * 2);
  bf16_t* Ctx = (bf16_t*)carve(nQ * 2);

  const dim3 blk(256);
  auto cvt = [&](const float* s, bf16_t* d, size_t n) {
    cvt_f32_to_bf16_kernel<<<dim3((unsigned)((n + 255) / 256)), blk, 0, stream>>>(s, d, (int)n);
  };
  cvt(hs, Xb, nX);
  cvt(Wq, Wqb, (size_t)Cc * Hc * Dc);
  cvt(Wk, Wkb, (size_t)Cc * HKVc * Dc);
  cvt(Wv, Wvb, (size_t)Cc * HKVc * Dc);
  cvt(Wo, Wob, (size_t)Hc * Dc * Cc);

  const int M = Bc * Tc;
  gemm_bf16_kernel<false><<<dim3((Hc * Dc) / 64, M / 64), blk, 0, stream>>>(
      Xb, Wqb, Qb, M, Hc * Dc, Cc);
  gemm_bf16_kernel<false><<<dim3((HKVc * Dc) / 64, M / 64), blk, 0, stream>>>(
      Xb, Wkb, Kbf, M, HKVc * Dc, Cc);
  gemm_bf16_kernel<false><<<dim3((HKVc * Dc) / 64, M / 64), blk, 0, stream>>>(
      Xb, Wvb, Vbf, M, HKVc * Dc, Cc);

  // fused scores + softmax + prune + renormalize (single attn write)
  attn_fused_kernel<<<dim3(Tc / 32, Bc * Hc), blk, 0, stream>>>(Qb, Kbf, keep, attn);

  context_kernel<<<dim3(1, Tc / 64, Bc * Hc), blk, 0, stream>>>(attn, Vbf, Ctx);

  gemm_bf16_kernel<true><<<dim3(Cc / 64, M / 64), blk, 0, stream>>>(
      Ctx, Wob, out, M, Cc, Hc * Dc);
}
